// MultiheadAttentionPooling_81982335746169
// MI455X (gfx1250) — compile-verified, matrix operand order checked by reading, not run on hardware
//
#include <hip/hip_runtime.h>
#include <cstdint>
#include <cstddef>

// ---------------------------------------------------------------------------
// MI455X (gfx1250) implementation: wave32, WMMA bf16 16x16x32 with f32 acc.
// fp32 precision recovered via bf16 hi/lo split (3 WMMAs per MAC tile).
// ---------------------------------------------------------------------------

#define DEVI __device__ __forceinline__

typedef __attribute__((ext_vector_type(16))) __bf16 v16bf;
typedef __attribute__((ext_vector_type(8)))  float  v8f;
typedef unsigned short u16;

struct alignas(16) U128 { unsigned int a, b, c, d; };
struct alignas(16) F4   { float v[4]; };
struct alignas(8)  S4   { u16 v[4]; };

union Frag { v16bf v; U128 q[2]; };

static constexpr int   Nn   = 32768;
static constexpr int   Dd   = 512;
static constexpr int   Bb   = 256;
static constexpr int   NPG  = 128;           // nodes per graph (contiguous)
static constexpr float EPS  = 1e-5f;

// ---- workspace layout (bytes) ---------------------------------------------
static constexpr size_t OFF_QVEC = 0;                         // 512 f32
static constexpr size_t OFF_QKB  = 4096;                      // 8 f32
static constexpr size_t OFF_QKW  = 8192;                      // 8*512 f32
static constexpr size_t OFF_WVHI = 32768;                     // 512*512 u16
static constexpr size_t OFF_WVLO = OFF_WVHI + 524288;
static constexpr size_t OFF_W1HI = OFF_WVLO + 524288;
static constexpr size_t OFF_W1LO = OFF_W1HI + 524288;
static constexpr size_t OFF_W2HI = OFF_W1LO + 524288;
static constexpr size_t OFF_W2LO = OFF_W2HI + 524288;
static constexpr size_t OFF_ATTN = OFF_W2LO + 524288;         // N*8 f32
static constexpr size_t OFF_AHI  = OFF_ATTN + (size_t)Nn*8*4; // N*512 u16
static constexpr size_t OFF_ALO  = OFF_AHI + (size_t)Nn*Dd*2;
static constexpr size_t OFF_BUFB = OFF_ALO + (size_t)Nn*Dd*2; // N*512 f32 (reused as H1 hi/lo)

// ---- bf16 helpers (round-to-nearest-even split) ---------------------------
DEVI u16 f2bf(float f) {
  unsigned int u = __float_as_uint(f);
  unsigned int r = u + 0x7FFFu + ((u >> 16) & 1u);
  return (u16)(r >> 16);
}
DEVI float bf2f(u16 h) { return __uint_as_float(((unsigned int)h) << 16); }

DEVI v8f wmma_bf16(const Frag& a, const Frag& b, v8f c) {
  return __builtin_amdgcn_wmma_f32_16x16x32_bf16(false, a.v, false, b.v,
                                                 (short)0, c, false, false);
}

// ===========================================================================
// K1: Qvec = seed @ WQ.T + bQ ; qkb[h] = (1/8) * sum_{j in head} Q[j]*bK[j]
// ===========================================================================
__global__ void k_qvec(const float* __restrict__ seed, const float* __restrict__ WQ,
                       const float* __restrict__ bQ, const float* __restrict__ bK,
                       float* __restrict__ Qvec, float* __restrict__ qkb) {
  __shared__ float qv[512];
  int t = threadIdx.x;
  for (int j = t; j < 512; j += 256) {
    float s = bQ[j];
    const float* row = WQ + (size_t)j * 512;
    for (int i = 0; i < 512; ++i) s = fmaf(seed[i], row[i], s);
    qv[j] = s;
    Qvec[j] = s;
  }
  __syncthreads();
  if (t < 8) {
    float s = 0.f;
    for (int j = 0; j < 64; ++j) s = fmaf(qv[t * 64 + j], bK[t * 64 + j], s);
    qkb[t] = 0.125f * s;
  }
}

// ===========================================================================
// K2: qkw[h,i] = (1/8) * sum_{j=0..63} Qvec[64h+j] * WK[64h+j, i]
// ===========================================================================
__global__ void k_qkw(const float* __restrict__ Qvec, const float* __restrict__ WK,
                      float* __restrict__ qkw) {
  int p = blockIdx.x * 256 + threadIdx.x;   // 0..4095
  int h = p >> 9, i = p & 511;
  float s = 0.f;
  for (int j = 0; j < 64; ++j)
    s = fmaf(Qvec[h * 64 + j], WK[(size_t)(h * 64 + j) * 512 + i], s);
  qkw[p] = 0.125f * s;   // fold 1/sqrt(64)
}

// ===========================================================================
// K3: fp32 -> bf16 hi/lo split for WV, W1, W2
// ===========================================================================
__global__ void k_cvtw(const float* __restrict__ WV, const float* __restrict__ W1,
                       const float* __restrict__ W2,
                       u16* __restrict__ WVhi, u16* __restrict__ WVlo,
                       u16* __restrict__ W1hi, u16* __restrict__ W1lo,
                       u16* __restrict__ W2hi, u16* __restrict__ W2lo) {
  int id  = blockIdx.x * 256 + threadIdx.x;     // 0..196607
  int mat = id >> 16;
  int e   = (id & 65535) * 4;
  const float* src = (mat == 0) ? WV : ((mat == 1) ? W1 : W2);
  u16* dhi = (mat == 0) ? WVhi : ((mat == 1) ? W1hi : W2hi);
  u16* dlo = (mat == 0) ? WVlo : ((mat == 1) ? W1lo : W2lo);
  F4 v = *(const F4*)&src[e];
  S4 hi, lo;
#pragma unroll
  for (int k = 0; k < 4; ++k) {
    hi.v[k] = f2bf(v.v[k]);
    lo.v[k] = f2bf(v.v[k] - bf2f(hi.v[k]));
  }
  *(S4*)&dhi[e] = hi;
  *(S4*)&dlo[e] = lo;
}

// ===========================================================================
// K4: per-graph GraphNorm(x) -> KV (bf16 hi/lo) + attention logits + softmax.
// One workgroup per graph (128 contiguous rows). LDS holds hi-bf16 KV tile.
// ===========================================================================
__global__ void k_gn_attn(const float* __restrict__ x,
                          const float* __restrict__ gw, const float* __restrict__ gb,
                          const float* __restrict__ gms,
                          const float* __restrict__ qkw, const float* __restrict__ qkb,
                          u16* __restrict__ KVhi, u16* __restrict__ KVlo,
                          float* __restrict__ attnw) {
  extern __shared__ float smemF[];
  float* qkwL   = smemF;                 // 4096 f32
  float* logits = smemF + 4096;          // 1024 f32
  float* hred   = smemF + 5120;          // 16 f32
  u16*   kvt    = (u16*)(smemF + 5136);  // 128 * 520 u16

  const int t = threadIdx.x;
  const int g = blockIdx.x;
  const float* base = x + (size_t)g * NPG * Dd;

  for (int i = t; i < 4096; i += 256) qkwL[i] = qkw[i];

  const int c1 = t, c2 = t + 256;
  // mean
  float s1 = 0.f, s2 = 0.f;
  for (int r = 0; r < NPG; ++r) {
    s1 += base[(size_t)r * Dd + c1];
    s2 += base[(size_t)r * Dd + c2];
  }
  float msm1 = (s1 * (1.f / NPG)) * gms[c1];
  float msm2 = (s2 * (1.f / NPG)) * gms[c2];
  // variance of (x - mean*ms)
  float v1 = 0.f, v2 = 0.f;
  for (int r = 0; r < NPG; ++r) {
    float d1 = base[(size_t)r * Dd + c1] - msm1;
    float d2 = base[(size_t)r * Dd + c2] - msm2;
    v1 = fmaf(d1, d1, v1); v2 = fmaf(d2, d2, v2);
  }
  float scl1 = gw[c1] * rsqrtf(v1 * (1.f / NPG) + EPS);
  float scl2 = gw[c2] * rsqrtf(v2 * (1.f / NPG) + EPS);
  float shf1 = gb[c1], shf2 = gb[c2];
  // normalize, write bf16 hi/lo to global, keep hi tile in LDS
  for (int r = 0; r < NPG; ++r) {
    size_t gi = (size_t)(g * NPG + r) * Dd;
    float d1 = fmaf(base[(size_t)r * Dd + c1] - msm1, scl1, shf1);
    float d2 = fmaf(base[(size_t)r * Dd + c2] - msm2, scl2, shf2);
    u16 h1 = f2bf(d1), h2 = f2bf(d2);
    KVhi[gi + c1] = h1;                 KVhi[gi + c2] = h2;
    KVlo[gi + c1] = f2bf(d1 - bf2f(h1)); KVlo[gi + c2] = f2bf(d2 - bf2f(h2));
    kvt[r * 520 + c1] = h1;             kvt[r * 520 + c2] = h2;
  }
  __syncthreads();
  // logits[row][head] = qkb[head] + KV[row,:] . qkw[head,:]
#pragma unroll
  for (int k = 0; k < 4; ++k) {
    int p = t + 256 * k;                // 0..1023
    int row = p >> 3, head = p & 7;
    float s = qkb[head];
    const float* qrow = qkwL + head * 512;
    const u16*   krow = kvt + row * 520;
    for (int i = 0; i < 512; ++i) s = fmaf(bf2f(krow[i]), qrow[i], s);
    logits[p] = s;
  }
  __syncthreads();
  if (t < 8) {                          // per-head max & exp-sum over 128 rows
    float m = -1e30f;
    for (int r = 0; r < NPG; ++r) m = fmaxf(m, logits[r * 8 + t]);
    float ss = 0.f;
    for (int r = 0; r < NPG; ++r) ss += __expf(logits[r * 8 + t] - m);
    hred[t] = m; hred[8 + t] = ss;
  }
  __syncthreads();
#pragma unroll
  for (int k = 0; k < 4; ++k) {
    int p = t + 256 * k;
    int row = p >> 3, head = p & 7;
    attnw[(size_t)(g * NPG + row) * 8 + head] =
        __expf(logits[p] - hred[head]) / hred[8 + head];
  }
}

// ===========================================================================
// K6: per-graph GraphNorm(weighted_V) -> nV (bf16 hi/lo); d_out = mean(wV)
// ===========================================================================
__global__ void k_gn2(const float* __restrict__ wv,
                      const float* __restrict__ gw, const float* __restrict__ gb,
                      const float* __restrict__ gms,
                      u16* __restrict__ nvhi, u16* __restrict__ nvlo,
                      float* __restrict__ outMean) {
  const int t = threadIdx.x, g = blockIdx.x;
  const float* base = wv + (size_t)g * NPG * Dd;
  const int c1 = t, c2 = t + 256;
  float s1 = 0.f, s2 = 0.f;
  for (int r = 0; r < NPG; ++r) {
    s1 += base[(size_t)r * Dd + c1];
    s2 += base[(size_t)r * Dd + c2];
  }
  float m1 = s1 * (1.f / NPG), m2 = s2 * (1.f / NPG);
  outMean[(size_t)g * Dd + c1] = m1;    // final out = mean(wV) + mean(mlp)
  outMean[(size_t)g * Dd + c2] = m2;
  float msm1 = m1 * gms[c1], msm2 = m2 * gms[c2];
  float v1 = 0.f, v2 = 0.f;
  for (int r = 0; r < NPG; ++r) {
    float d1 = base[(size_t)r * Dd + c1] - msm1;
    float d2 = base[(size_t)r * Dd + c2] - msm2;
    v1 = fmaf(d1, d1, v1); v2 = fmaf(d2, d2, v2);
  }
  float scl1 = gw[c1] * rsqrtf(v1 * (1.f / NPG) + EPS);
  float scl2 = gw[c2] * rsqrtf(v2 * (1.f / NPG) + EPS);
  float shf1 = gb[c1], shf2 = gb[c2];
  for (int r = 0; r < NPG; ++r) {
    size_t gi = (size_t)(g * NPG + r) * Dd;
    float d1 = fmaf(base[(size_t)r * Dd + c1] - msm1, scl1, shf1);
    float d2 = fmaf(base[(size_t)r * Dd + c2] - msm2, scl2, shf2);
    u16 h1 = f2bf(d1), h2 = f2bf(d2);
    nvhi[gi + c1] = h1;                 nvhi[gi + c2] = h2;
    nvlo[gi + c1] = f2bf(d1 - bf2f(h1)); nvlo[gi + c2] = f2bf(d2 - bf2f(h2));
  }
}

// ===========================================================================
// WMMA GEMM: C[128x128] = A[128x512] @ W[128rows x 512].T with split-bf16.
// 8 waves, each 32x64 (2x4 tiles of 16x16). Fused epilogues:
//   mode 0: out = (C + bias) * attn[row, col/64] + resid   (weighted V)
//   mode 1: gelu(C + bias) -> bf16 hi/lo                    (MLP hidden)
//   mode 2: d_out[graph, col] += bias + mean_rows(C)        (pooled MLP out)
// ===========================================================================
__global__ __launch_bounds__(256)
void gemm_fused(const u16* __restrict__ Ahi, const u16* __restrict__ Alo,
                const u16* __restrict__ Whi, const u16* __restrict__ Wlo,
                const float* __restrict__ bias,
                const float* __restrict__ attn, const float* __restrict__ resid,
                float* __restrict__ outF,
                u16* __restrict__ outHi, u16* __restrict__ outLo,
                float* __restrict__ poolOut, int mode) {
  __shared__ u16 sAhi[128 * 40];
  __shared__ u16 sAlo[128 * 40];
  __shared__ u16 sBhi[128 * 40];
  __shared__ u16 sBlo[128 * 40];
  __shared__ float pbuf[4][128];

  const int t = threadIdx.x;
  const int lane = t & 31, wm = (t >> 5) & 3, wn = t >> 7;
  const int m0 = blockIdx.y * 128, n0 = blockIdx.x * 128;

  v8f acc[2][4] = {};
  const int r  = t >> 1;
  const int hh = (t & 1) << 4;          // 0 or 16 halves
  const int kc = ((lane >> 4) & 1) << 3; // fragment k-offset: 0 or 8 halves

  for (int ks = 0; ks < 16; ++ks) {
    const int k0 = ks * 32;
    // stage tiles to LDS (row stride 40 halves: conflict-free b128 reads)
    const u16* pAh = Ahi + (size_t)(m0 + r) * 512 + k0 + hh;
    const u16* pAl = Alo + (size_t)(m0 + r) * 512 + k0 + hh;
    const u16* pBh = Whi + (size_t)(n0 + r) * 512 + k0 + hh;
    const u16* pBl = Wlo + (size_t)(n0 + r) * 512 + k0 + hh;
    *(U128*)&sAhi[r * 40 + hh]     = *(const U128*)(pAh);
    *(U128*)&sAhi[r * 40 + hh + 8] = *(const U128*)(pAh + 8);
    *(U128*)&sAlo[r * 40 + hh]     = *(const U128*)(pAl);
    *(U128*)&sAlo[r * 40 + hh + 8] = *(const U128*)(pAl + 8);
    *(U128*)&sBhi[r * 40 + hh]     = *(const U128*)(pBh);
    *(U128*)&sBhi[r * 40 + hh + 8] = *(const U128*)(pBh + 8);
    *(U128*)&sBlo[r * 40 + hh]     = *(const U128*)(pBl);
    *(U128*)&sBlo[r * 40 + hh + 8] = *(const U128*)(pBl + 8);
    if (ks + 1 < 16) {                  // pull next k-tile toward the caches
      __builtin_prefetch(pAh + 32, 0, 1);
      __builtin_prefetch(pBh + 32, 0, 1);
    }
    __syncthreads();

    Frag fAh[2], fAl[2], fBh[4], fBl[4];
#pragma unroll
    for (int mt = 0; mt < 2; ++mt) {
      int ra = (wm * 32 + mt * 16 + (lane & 15)) * 40 + kc;
      fAh[mt].q[0] = *(const U128*)&sAhi[ra];
      fAh[mt].q[1] = *(const U128*)&sAhi[ra + 16];
      fAl[mt].q[0] = *(const U128*)&sAlo[ra];
      fAl[mt].q[1] = *(const U128*)&sAlo[ra + 16];
    }
#pragma unroll
    for (int nt = 0; nt < 4; ++nt) {
      int rb = (wn * 64 + nt * 16 + (lane & 15)) * 40 + kc;
      fBh[nt].q[0] = *(const U128*)&sBhi[rb];
      fBh[nt].q[1] = *(const U128*)&sBhi[rb + 16];
      fBl[nt].q[0] = *(const U128*)&sBlo[rb];
      fBl[nt].q[1] = *(const U128*)&sBlo[rb + 16];
    }
#pragma unroll
    for (int mt = 0; mt < 2; ++mt)
#pragma unroll
      for (int nt = 0; nt < 4; ++nt) {
        acc[mt][nt] = wmma_bf16(fAl[mt], fBh[nt], acc[mt][nt]);
        acc[mt][nt] = wmma_bf16(fAh[mt], fBl[nt], acc[mt][nt]);
        acc[mt][nt] = wmma_bf16(fAh[mt], fBh[nt], acc[mt][nt]);
      }
    __syncthreads();
  }

  // ---- epilogues ----------------------------------------------------------
  if (mode == 0) {
#pragma unroll
    for (int mt = 0; mt < 2; ++mt)
#pragma unroll
      for (int nt = 0; nt < 4; ++nt)
#pragma unroll
        for (int gg = 0; gg < 8; ++gg) {
          int row = m0 + wm * 32 + mt * 16 + gg + ((lane >> 4) << 3);
          int col = n0 + wn * 64 + nt * 16 + (lane & 15);
          float c = acc[mt][nt][gg] + bias[col];
          outF[(size_t)row * Dd + col] =
              fmaf(c, attn[(size_t)row * 8 + (col >> 6)],
                   resid[(size_t)row * Dd + col]);
        }
  } else if (mode == 1) {
#pragma unroll
    for (int mt = 0; mt < 2; ++mt)
#pragma unroll
      for (int nt = 0; nt < 4; ++nt)
#pragma unroll
        for (int gg = 0; gg < 8; ++gg) {
          int row = m0 + wm * 32 + mt * 16 + gg + ((lane >> 4) << 3);
          int col = n0 + wn * 64 + nt * 16 + (lane & 15);
          float c = acc[mt][nt][gg] + bias[col];
          float ge = 0.5f * c * (1.0f + erff(c * 0.70710678118654752f));
          u16 hv = f2bf(ge);
          size_t idx = (size_t)row * Dd + col;
          outHi[idx] = hv;
          outLo[idx] = f2bf(ge - bf2f(hv));
        }
  } else {
    // pooled mean over the 128 rows of this tile (== one graph)
    float ps[4];
#pragma unroll
    for (int nt = 0; nt < 4; ++nt) {
      float s = 0.f;
#pragma unroll
      for (int mt = 0; mt < 2; ++mt)
#pragma unroll
        for (int gg = 0; gg < 8; ++gg) s += acc[mt][nt][gg];
      s += __shfl_xor(s, 16, 32);       // fold lanes 16-31 (rows +8)
      ps[nt] = s;
    }
    if (lane < 16) {
#pragma unroll
      for (int nt = 0; nt < 4; ++nt)
        pbuf[wm][wn * 64 + nt * 16 + lane] = ps[nt];
    }
    __syncthreads();
    if (t < 128) {
      float s = pbuf[0][t] + pbuf[1][t] + pbuf[2][t] + pbuf[3][t];
      int gc = n0 + t;
      size_t oi = (size_t)blockIdx.y * Dd + gc;   // blockIdx.y == graph id
      poolOut[oi] += bias[gc] + s * (1.0f / NPG);
    }
  }
}

// ===========================================================================
extern "C" void kernel_launch(void* const* d_in, const int* in_sizes, int n_in,
                              void* d_out, int out_size, void* d_ws, size_t ws_size,
                              hipStream_t stream) {
  const float* x     = (const float*)d_in[0];
  const float* seed  = (const float*)d_in[3];
  const float* WQ    = (const float*)d_in[4];
  const float* bQ    = (const float*)d_in[5];
  const float* WK    = (const float*)d_in[6];
  const float* bK    = (const float*)d_in[7];
  const float* WV    = (const float*)d_in[8];
  const float* bV    = (const float*)d_in[9];
  const float* W1    = (const float*)d_in[10];
  const float* b1    = (const float*)d_in[11];
  const float* W2    = (const float*)d_in[12];
  const float* b2    = (const float*)d_in[13];
  const float* nkv_w = (const float*)d_in[14];
  const float* nkv_b = (const float*)d_in[15];
  const float* nkv_ms= (const float*)d_in[16];
  const float* no_w  = (const float*)d_in[17];
  const float* no_b  = (const float*)d_in[18];
  const float* no_ms = (const float*)d_in[19];
  float* out = (float*)d_out;

  char* ws = (char*)d_ws;
  float* Qvec = (float*)(ws + OFF_QVEC);
  float* qkb  = (float*)(ws + OFF_QKB);
  float* qkw  = (float*)(ws + OFF_QKW);
  u16* WVhi = (u16*)(ws + OFF_WVHI);
  u16* WVlo = (u16*)(ws + OFF_WVLO);
  u16* W1hi = (u16*)(ws + OFF_W1HI);
  u16* W1lo = (u16*)(ws + OFF_W1LO);
  u16* W2hi = (u16*)(ws + OFF_W2HI);
  u16* W2lo = (u16*)(ws + OFF_W2LO);
  float* attnw = (float*)(ws + OFF_ATTN);
  u16* Ahi = (u16*)(ws + OFF_AHI);      // KV hi, later nV hi
  u16* Alo = (u16*)(ws + OFF_ALO);      // KV lo, later nV lo
  float* bufB = (float*)(ws + OFF_BUFB);// weighted_V, later H1 hi/lo
  u16* H1hi = (u16*)(ws + OFF_BUFB);
  u16* H1lo = H1hi + (size_t)Nn * Dd;

  const size_t SMEM_GN = (size_t)(4096 + 1024 + 16) * 4 + (size_t)128 * 520 * 2;

  k_qvec<<<1, 256, 0, stream>>>(seed, WQ, bQ, bK, Qvec, qkb);
  k_qkw<<<16, 256, 0, stream>>>(Qvec, WK, qkw);
  k_cvtw<<<768, 256, 0, stream>>>(WV, W1, W2, WVhi, WVlo, W1hi, W1lo, W2hi, W2lo);
  k_gn_attn<<<Bb, 256, SMEM_GN, stream>>>(x, nkv_w, nkv_b, nkv_ms, qkw, qkb,
                                          Ahi, Alo, attnw);
  // weighted_V = (KV @ WV.T + bV) * attn + x
  gemm_fused<<<dim3(4, 256), 256, 0, stream>>>(Ahi, Alo, WVhi, WVlo, bV,
                                               attnw, x, bufB,
                                               nullptr, nullptr, nullptr, 0);
  // nV = graphnorm(weighted_V); out = mean(weighted_V)
  k_gn2<<<Bb, 256, 0, stream>>>(bufB, no_w, no_b, no_ms, Ahi, Alo, out);
  // H1 = gelu(nV @ W1.T + b1) -> bf16 hi/lo
  gemm_fused<<<dim3(4, 256), 256, 0, stream>>>(Ahi, Alo, W1hi, W1lo, b1,
                                               nullptr, nullptr, nullptr,
                                               H1hi, H1lo, nullptr, 1);
  // out += b2 + mean_rows(H1 @ W2.T)
  gemm_fused<<<dim3(4, 256), 256, 0, stream>>>(H1hi, H1lo, W2hi, W2lo, b2,
                                               nullptr, nullptr, nullptr,
                                               nullptr, nullptr, out, 2);
}
